// LASEREmbedderI_8787503088338
// MI455X (gfx1250) — compile-verified, matrix operand order checked by reading, not run on hardware
//
#include <hip/hip_runtime.h>
#include <math.h>

// ---------------- problem constants ----------------
#define T_SEQ  256
#define BATCH  128
#define EMB    320
#define HID    512
#define GD     2048   // 4*H, gate dim
#define NLAYER 5
#define HPAD   520    // padded LDS row stride for h slab, in bf16 (1040 B)

typedef __attribute__((ext_vector_type(16))) __bf16 v16bf;
typedef __attribute__((ext_vector_type(8)))  float  v8f;

// ---------------------------------------------------------------------------
// WMMA fragment loaders (wave32, v_wmma_f32_16x16x32_bf16)
//
// A (16x32 bf16, M x K), documented layout:
//   lanes 0-15 : M=lane,    VGPR0..3 = K 0..7 (pairs), VGPR4..7 = K 16..23
//   lanes16-31 : M=lane-16, VGPR0..3 = K 8..15,        VGPR4..7 = K 24..31
// -> per lane: two contiguous 16-byte loads from a row-major bf16 matrix.
//
// B (32x16 bf16, K x N): weights are PRE-PACKED (pack_w kernel below) into
// tile-lane-major order so each lane reads 16 contiguous bf16 (32 bytes).
//
// C/D (16x16 f32), documented layout:
//   VGPR r, lanes 0-15 -> (M=r,   N=lane); lanes 16-31 -> (M=8+r, N=lane-16)
// ---------------------------------------------------------------------------
static __device__ __forceinline__ v16bf load_a_frag(const __bf16* __restrict__ X,
                                                    int ldk, int m0, int k0, int lane) {
  int row = m0 + (lane & 15);
  int kh  = (lane >> 4) << 3;              // 0 or 8
  const __bf16* p = X + (size_t)row * ldk + (k0 + kh);
  union { uint4 u[2]; v16bf v; } u;
  u.u[0] = *(const uint4*)p;               // k0+kh .. k0+kh+7
  u.u[1] = *(const uint4*)(p + 16);        // k0+kh+16 .. k0+kh+23
  return u.v;
}

static __device__ __forceinline__ v16bf load_b_frag(const __bf16* __restrict__ Wp,
                                                    int ntile, int ktile, int ktiles,
                                                    int lane) {
  const __bf16* p = Wp + (((size_t)ntile * ktiles + ktile) << 9) + (lane << 4);
  union { uint4 u[2]; v16bf v; } u;
  u.u[0] = *(const uint4*)p;
  u.u[1] = *(const uint4*)(p + 8);
  return u.v;
}

#define WMMA_BF16(A, B, C) \
  __builtin_amdgcn_wmma_f32_16x16x32_bf16(false, (A), false, (B), (short)0, (C), false, false)

// ---------------------------------------------------------------------------
// Embedding lookup + f32 -> bf16
// ---------------------------------------------------------------------------
__global__ void embed_bf16_kernel(const int* __restrict__ tokens,
                                  const float* __restrict__ embed,
                                  __bf16* __restrict__ x0, int n) {
  int idx = blockIdx.x * blockDim.x + threadIdx.x;
  if (idx >= n) return;
  int e  = idx % EMB;
  int tb = idx / EMB;                       // t*B + b (tokens is [T][B] row-major)
  int tok = tokens[tb];
  x0[idx] = (__bf16)embed[(size_t)tok * EMB + e];
}

// combined bias b_ih + b_hh, all layers/dirs -> bias[l][d][g] f32
__global__ void bias_kernel(const float* __restrict__ bi0, const float* __restrict__ bh0,
                            const float* __restrict__ bi,  const float* __restrict__ bh,
                            float* __restrict__ bias) {
  int idx = blockIdx.x * blockDim.x + threadIdx.x;   // NLAYER*2*GD
  if (idx >= NLAYER * 2 * GD) return;
  int g = idx % GD;
  int d = (idx / GD) & 1;
  int l = idx / (2 * GD);
  float v;
  if (l == 0) v = bi0[d * GD + g] + bh0[d * GD + g];
  else {
    size_t o = ((size_t)(l - 1) * 2 + d) * GD + g;
    v = bi[o] + bh[o];
  }
  bias[idx] = v;
}

// pack row-major f32 W[G][K] into WMMA-B bf16 tiles (lane-major, see load_b_frag)
__global__ void pack_w_kernel(const float* __restrict__ W, __bf16* __restrict__ Wp,
                              int G, int K) {
  int idx = blockIdx.x * blockDim.x + threadIdx.x;   // G*K
  if (idx >= G * K) return;
  int ktiles = K >> 5;
  int tid  = idx >> 9;            // 512 elems per 32x16 tile
  int e    = idx & 511;
  int lane = e >> 4, q = e & 15;
  int nt = tid / ktiles, kt = tid % ktiles;
  int n  = lane & 15;
  int kh = (lane >> 4) << 3;
  int k  = (q < 8) ? (kh + q) : (16 + kh + (q - 8));
  Wp[idx] = (__bf16)W[(size_t)(nt * 16 + n) * K + (kt << 5) + k];
}

__global__ void zero_hc_kernel(float* __restrict__ cst, __bf16* __restrict__ hst0) {
  int idx = blockIdx.x * blockDim.x + threadIdx.x;   // 2*BATCH*HID
  if (idx >= 2 * BATCH * HID) return;
  cst[idx] = 0.f;
  hst0[idx] = (__bf16)0.f;                           // parity-0 h buffer only
}

// ---------------------------------------------------------------------------
// Time-batched input projection: Gin[d][t*B+b][g] = X @ Wi^T + (bi+bh)
// Per wave: 32(M) x 64(N) tile, f32 accum. grid = ((M/32)*(GD/64)/8, 2 dirs)
// ---------------------------------------------------------------------------
__global__ void __launch_bounds__(256)
gates_gemm_kernel(const __bf16* __restrict__ X,
                  const __bf16* __restrict__ WpBase, size_t wpDirStride,
                  const float* __restrict__ biasL,
                  float* __restrict__ ginBase, size_t outDirStride,
                  int M, int K) {
  const int N = GD;
  const int dir = blockIdx.y;
  const __bf16* Wp   = WpBase + dir * wpDirStride;
  const float*  bias = biasL + dir * GD;
  float*        out  = ginBase + dir * outDirStride;

  int lane = threadIdx.x & 31, wid = threadIdx.x >> 5;
  int gw = blockIdx.x * 8 + wid;
  int mstrips = M >> 5;
  int ms = gw % mstrips, ns = gw / mstrips;
  int m0 = ms << 5, n0 = ns << 6;
  int ktiles = K >> 5;
  int col0 = lane & 15, hb = (lane >> 4) << 3;

  v8f acc[2][4];
  #pragma unroll
  for (int j = 0; j < 4; ++j) {
    float bv = bias[n0 + (j << 4) + col0];
    #pragma unroll
    for (int r = 0; r < 8; ++r) { acc[0][j][r] = bv; acc[1][j][r] = bv; }
  }

  for (int kt = 0; kt < ktiles; ++kt) {
    v16bf a0 = load_a_frag(X, K, m0, kt << 5, lane);
    v16bf a1 = load_a_frag(X, K, m0 + 16, kt << 5, lane);
    if (kt + 1 < ktiles)   // warm caches for next B tiles (global_prefetch_b8)
      __builtin_prefetch(Wp + (((size_t)(ns << 2) * ktiles + kt + 1) << 9), 0, 3);
    #pragma unroll
    for (int j = 0; j < 4; ++j) {
      v16bf b = load_b_frag(Wp, (ns << 2) + j, kt, ktiles, lane);
      acc[0][j] = WMMA_BF16(a0, b, acc[0][j]);
      acc[1][j] = WMMA_BF16(a1, b, acc[1][j]);
    }
  }

  #pragma unroll
  for (int i = 0; i < 2; ++i)
    #pragma unroll
    for (int j = 0; j < 4; ++j) {
      float* o = out + (size_t)(m0 + (i << 4) + hb) * N + n0 + (j << 4) + col0;
      #pragma unroll
      for (int r = 0; r < 8; ++r) o[(size_t)r * N] = acc[i][j][r];
    }
}

// ---------------------------------------------------------------------------
// FUSED recurrence step (one launch per t, both directions):
//   1) async-copy h_prev[dir] (128 KB) into LDS (global_load_async_to_lds_b128,
//      ASYNCcnt), overlapped with the Gin C-tile loads,
//   2) gates = Gin[time] + h_prev @ Wh^T with A-fragments fed from LDS
//      (ds_load_b128, short dependency chains on the serial critical path),
//   3) spill gates to LDS (reusing the h slab, they are time-disjoint),
//   4) LSTM pointwise; h double-buffered across launches.
//
// Block (j0 = 32*blockIdx.x, dir = blockIdx.y) owns gate columns
// { c*512 + j0 .. c*512+j0+31 } for all four gates c=0..3 and all B rows.
// LDS: max(h slab 128x520x2 = 130 KB, gate slab 128x128x4 = 64 KB) = 130 KB.
// h rows padded to 1040 B (bank rotation 4) -> conflict-free b128 A-frag reads.
// ---------------------------------------------------------------------------
__global__ void __launch_bounds__(256)
lstm_step_kernel(const __bf16* __restrict__ Hprev,   // [2][B][H] bf16 (parity t)
                 __bf16* __restrict__ Hnext,         // [2][B][H] bf16 (parity t+1)
                 const __bf16* __restrict__ WhL, size_t whDirStride,
                 const float* __restrict__ ginL, size_t ginDirStride,
                 float* __restrict__ cstate,         // [2][B][H] f32
                 __bf16* __restrict__ xnext,         // [T][B][2H] bf16
                 float* __restrict__ dout,           // [T][B][H] f32 (last layer)
                 int t, int isLast) {
  __shared__ __align__(16) unsigned char smem[BATCH * HPAD * 2];  // 130 KB

  const int dir  = blockIdx.y;
  const int j0   = blockIdx.x << 5;                  // 0..480
  const int time = dir ? (T_SEQ - 1 - t) : t;        // bwd consumes reversed time
  const __bf16* H  = Hprev + (size_t)dir * BATCH * HID;
  const __bf16* Wp = WhL + dir * whDirStride;
  const float*  C  = ginL + dir * ginDirStride + (size_t)time * BATCH * GD;

  const int tid  = threadIdx.x;
  const int lane = tid & 31, w = tid >> 5;
  const int ms = w & 3, nh = w >> 2;                 // 4 m-strips, 2 gate-pairs
  const int m0 = ms << 5;
  const int col0 = lane & 15, hb = (lane >> 4) << 3;
  const int ktiles = HID >> 5;                       // 16

  // ---- 1) bulk async copy of h_prev[dir] into padded LDS slab ----
  {
    unsigned ldsbase = (unsigned)(uintptr_t)(void*)smem;
    #pragma unroll
    for (int i = 0; i < 32; ++i) {                   // 8192 x 16B chunks total
      int ch  = (i << 8) + tid;                      // chunk id
      int el  = ch << 3;                             // bf16 element index
      int row = el >> 9, col = el & (HID - 1);
      unsigned loff = ldsbase + (unsigned)(row * (HPAD * 2) + (col << 1));
      const __bf16* gp = H + el;
      asm volatile("global_load_async_to_lds_b128 %0, %1, off"
                   :: "v"(loff), "v"(gp) : "memory");
    }
  }

  // ---- GEMM C-init from Gin while the async copy is in flight ----
  v8f acc[2][4];
  #pragma unroll
  for (int i = 0; i < 2; ++i)
    #pragma unroll
    for (int j = 0; j < 4; ++j) {
      int c  = (nh << 1) + (j >> 1);                 // gate chunk 0..3
      int jt = j & 1;                                // which 16-col tile of the 32
      int gcol = c * HID + j0 + (jt << 4) + col0;    // global gate column
      #pragma unroll
      for (int r = 0; r < 8; ++r)
        acc[i][j][r] = C[(size_t)(m0 + (i << 4) + hb + r) * GD + gcol];
    }

  asm volatile("s_wait_asynccnt 0x0" ::: "memory");  // own wave's async done
  __syncthreads();                                   // all waves' async visible

  // ---- 2) k-loop: A-fragments from LDS, B-fragments from packed weights ----
  const __bf16* hsh = (const __bf16*)smem;
  for (int kt = 0; kt < ktiles; ++kt) {
    v16bf a0, a1;
    {
      int row = m0 + (lane & 15);
      int kh  = (lane >> 4) << 3;
      const __bf16* p0 = hsh + row * HPAD + (kt << 5) + kh;
      const __bf16* p1 = p0 + 16 * HPAD;
      union { uint4 u[2]; v16bf v; } ua, ub;
      ua.u[0] = *(const uint4*)p0; ua.u[1] = *(const uint4*)(p0 + 16);
      ub.u[0] = *(const uint4*)p1; ub.u[1] = *(const uint4*)(p1 + 16);
      a0 = ua.v; a1 = ub.v;
    }
    #pragma unroll
    for (int j = 0; j < 4; ++j) {
      int c  = (nh << 1) + (j >> 1);
      int jt = j & 1;
      int ntile = (c << 5) + (j0 >> 4) + jt;         // B-tile index in [0,128)
      v16bf b = load_b_frag(Wp, ntile, kt, ktiles, lane);
      acc[0][j] = WMMA_BF16(a0, b, acc[0][j]);
      acc[1][j] = WMMA_BF16(a1, b, acc[1][j]);
    }
  }

  __syncthreads();                                   // h slab dead -> reuse as gates

  // ---- 3) spill gates to LDS: gl[b*128 + c*32 + j'] ----
  float* gl = (float*)smem;
  #pragma unroll
  for (int i = 0; i < 2; ++i)
    #pragma unroll
    for (int j = 0; j < 4; ++j) {
      int c  = (nh << 1) + (j >> 1);
      int jt = j & 1;
      int cinb = (c << 5) + (jt << 4) + col0;        // column within slab
      #pragma unroll
      for (int r = 0; r < 8; ++r)
        gl[(m0 + (i << 4) + hb + r) * 128 + cinb] = acc[i][j][r];
    }

  __syncthreads();

  // ---- 4) LSTM pointwise for this block's (b, j0+j') elements ----
  const int trow = time;                              // output row for this dir
  #pragma unroll
  for (int e = tid; e < BATCH * 32; e += 256) {
    int b  = e >> 5;
    int jp = e & 31;
    const float* gb = gl + b * 128 + jp;
    float gi = gb[0], gf = gb[32], gg = gb[64], go = gb[96];
    int j = j0 + jp;
    int cidx = ((dir * BATCH) + b) * HID + j;
    float c = cstate[cidx];
    float si = 1.f / (1.f + expf(-gi));
    float sf = 1.f / (1.f + expf(-gf));
    float so = 1.f / (1.f + expf(-go));
    float cn = sf * c + si * tanhf(gg);
    float hn = so * tanhf(cn);
    cstate[cidx] = cn;
    Hnext[cidx] = (__bf16)hn;
    xnext[((size_t)trow * BATCH + b) * (2 * HID) + dir * HID + j] = (__bf16)hn;
    if (isLast) {
      // row r of dout first touched at launch min(r, T-1-r) => t<T/2 == first
      float* o = dout + ((size_t)trow * BATCH + b) * HID + j;
      if (t < T_SEQ / 2) *o = hn;
      else               *o = fmaxf(*o, hn);
    }
  }
}

// ---------------------------------------------------------------------------
// host side
// ---------------------------------------------------------------------------
extern "C" void kernel_launch(void* const* d_in, const int* in_sizes, int n_in,
                              void* d_out, int out_size, void* d_ws, size_t ws_size,
                              hipStream_t stream) {
  (void)in_sizes; (void)n_in; (void)out_size; (void)ws_size;
  const int*   tokens = (const int*)  d_in[0];
  const float* embedW = (const float*)d_in[1];
  const float* W_ih0  = (const float*)d_in[2];
  const float* W_hh0  = (const float*)d_in[3];
  const float* b_ih0  = (const float*)d_in[4];
  const float* b_hh0  = (const float*)d_in[5];
  const float* W_ih   = (const float*)d_in[6];
  const float* W_hh   = (const float*)d_in[7];
  const float* b_ih   = (const float*)d_in[8];
  const float* b_hh   = (const float*)d_in[9];
  float* out = (float*)d_out;

  // ---- workspace carve-up (~752 MB) ----
  char* w = (char*)d_ws; size_t off = 0;
  auto carve = [&](size_t bytes) { void* p = w + off; off = (off + bytes + 255) & ~(size_t)255; return p; };
  const size_t M = (size_t)T_SEQ * BATCH;                    // 32768
  const size_t HS = (size_t)2 * BATCH * HID;                 // one h/c parity
  __bf16* x0    = (__bf16*)carve(M * EMB * 2);
  __bf16* xbufA = (__bf16*)carve(M * 2 * HID * 2);
  __bf16* xbufB = (__bf16*)carve(M * 2 * HID * 2);
  __bf16* wip0  = (__bf16*)carve((size_t)2 * GD * EMB * 2);
  __bf16* wip   = (__bf16*)carve((size_t)4 * 2 * GD * (2 * HID) * 2);
  __bf16* whp   = (__bf16*)carve((size_t)NLAYER * 2 * GD * HID * 2);
  float*  bias  = (float*) carve((size_t)NLAYER * 2 * GD * 4);
  float*  gin   = (float*) carve((size_t)2 * M * GD * 4);
  float*  cst   = (float*) carve(HS * 4);
  __bf16* hst   = (__bf16*)carve(2 * HS * 2);                // double-buffered h

  // ---- one-time prep (cheap; part of every call for determinism) ----
  {
    int n = (int)(M * EMB);
    embed_bf16_kernel<<<(n + 255) / 256, 256, 0, stream>>>(tokens, embedW, x0, n);
  }
  bias_kernel<<<(NLAYER * 2 * GD + 255) / 256, 256, 0, stream>>>(b_ih0, b_hh0, b_ih, b_hh, bias);

  for (int d = 0; d < 2; ++d) {     // layer-0 input weights (K=EMB)
    int n = GD * EMB;
    pack_w_kernel<<<(n + 255) / 256, 256, 0, stream>>>(
        W_ih0 + (size_t)d * GD * EMB, wip0 + (size_t)d * GD * EMB, GD, EMB);
  }
  for (int l = 0; l < NLAYER; ++l)  // hidden weights (K=HID), all layers
    for (int d = 0; d < 2; ++d) {
      const float* src = (l == 0) ? (W_hh0 + (size_t)d * GD * HID)
                                  : (W_hh + ((size_t)(l - 1) * 2 + d) * GD * HID);
      int n = GD * HID;
      pack_w_kernel<<<(n + 255) / 256, 256, 0, stream>>>(
          src, whp + ((size_t)l * 2 + d) * GD * HID, GD, HID);
    }
  for (int l = 1; l < NLAYER; ++l)  // input weights layers 1..4 (K=2H)
    for (int d = 0; d < 2; ++d) {
      size_t o = ((size_t)(l - 1) * 2 + d) * GD * (2 * HID);
      int n = GD * 2 * HID;
      pack_w_kernel<<<(n + 255) / 256, 256, 0, stream>>>(W_ih + o, wip + o, GD, 2 * HID);
    }

  // ---- layer loop ----
  for (int l = 0; l < NLAYER; ++l) {
    const __bf16* X = (l == 0) ? x0 : ((l & 1) ? xbufA : xbufB);
    __bf16* xnext   = (l & 1) ? xbufB : xbufA;
    int K = (l == 0) ? EMB : 2 * HID;
    const __bf16* WiBase = (l == 0) ? wip0 : (wip + (size_t)(l - 1) * 2 * GD * (2 * HID));
    size_t wiDirStride = (size_t)GD * K;

    zero_hc_kernel<<<(2 * BATCH * HID + 255) / 256, 256, 0, stream>>>(cst, hst);

    // Gin = X @ Wi^T + bias, all T at once, both dirs
    {
      dim3 grid((unsigned)(((int)M / 32) * (GD / 64) / 8), 2);
      gates_gemm_kernel<<<grid, 256, 0, stream>>>(
          X, WiBase, wiDirStride, bias + (size_t)l * 2 * GD,
          gin, M * GD, (int)M, K);
    }

    const __bf16* WhL = whp + (size_t)l * 2 * GD * HID;
    int isLast = (l == NLAYER - 1);
    for (int t = 0; t < T_SEQ; ++t) {
      const __bf16* Hp = hst + (size_t)(t & 1) * HS;
      __bf16*       Hn = hst + (size_t)((t + 1) & 1) * HS;
      lstm_step_kernel<<<dim3(HID / 32, 2), 256, 0, stream>>>(
          Hp, Hn, WhL, (size_t)GD * HID, gin, M * GD,
          cst, xnext, out, t, isLast);
    }
  }
}
